// SurnameClassifier_87840671137937
// MI455X (gfx1250) — matrix operand order repaired and checked by reading, not compiled, not run on hardware
//
#include <hip/hip_runtime.h>
#include <math.h>

// Problem dimensions (compile-time constants from the reference).
#define BB 1024
#define SS 64
#define EE 256
#define HH 512
#define VV 128
#define CC 32

typedef __attribute__((ext_vector_type(16))) _Float16 v16h;
typedef __attribute__((ext_vector_type(8)))  _Float16 v8h;
typedef __attribute__((ext_vector_type(8)))  float    v8f;

// ---------------------------------------------------------------------------
// Fragment builders for v_wmma_f32_16x16x32_f16 (wave32), f16 source data.
//
// A (16x32, MxK): lane l (hi = l>=16): row M = l%16
//   elements i: K = kbase + 16*hi + i -> 32B contiguous
//
// B (32x16, KxN), weights row-major [N, K] (we multiply by W^T):
//   lane l: column N = n_base + l%16, elements i: K = kbase + 16*hi + i
//   -> 32B contiguous from the f16 weight row.
//
// C/D (16x16 f32): lane l, vgpr r -> (M = r + 8*(l>=16), N = l%16).
// ---------------------------------------------------------------------------
__device__ __forceinline__ v16h load_a_h16(const _Float16* __restrict__ row,
                                           int kbase, int hi) {
  return *(const v16h*)(row + kbase + 16 * hi);
}

__device__ __forceinline__ v16h load_b_h16(const _Float16* __restrict__ wrow,
                                           int kbase, int hi) {
  return *(const v16h*)(wrow + kbase + 16 * hi);
}

__device__ __forceinline__ v8f wmma_f16(v16h a, v16h b, v8f c) {
  return __builtin_amdgcn_wmma_f32_16x16x32_f16(
      /*neg_a=*/false, a, /*neg_b=*/false, b,
      /*c_mod=*/(short)0, c, /*reuse_a=*/false, /*reuse_b=*/false);
}

// ---------------------------------------------------------------------------
// f32 -> f16 tensor conversion (weights/emb are loop-invariant).
// ---------------------------------------------------------------------------
__global__ void __launch_bounds__(256)
cvt_f32_to_f16_kernel(const float* __restrict__ src, _Float16* __restrict__ dst,
                      int n) {
  int i = (blockIdx.x * 256 + threadIdx.x) * 4;
  if (i + 3 < n) {
#pragma unroll
    for (int j = 0; j < 4; ++j) dst[i + j] = (_Float16)src[i + j];
  }
}

// bias2[n] = b_ih[n] + b_hh[n]  (folded into the input-projection table P).
__global__ void __launch_bounds__(256)
bias_fold_kernel(const float* __restrict__ b_ih, const float* __restrict__ b_hh,
                 float* __restrict__ bias2, int n) {
  int i = blockIdx.x * 256 + threadIdx.x;
  if (i < n) bias2[i] = b_ih[i] + b_hh[i];
}

// ---------------------------------------------------------------------------
// Generic f16-operand WMMA GEMM: out = act(A[M,K] @ W[N,K]^T + bias[N]).
// Each wave owns a 16x32 tile; grid waves = (M/16)*(N/32).
// ACT: 0 = identity, 1 = relu.  OutT: _Float16 or float.
// Used for: P = emb @ W_ih^T + bias2  (M=V=128, N=H, K=E)   -> f32
//           hidden = relu(last @ W1^T + b1)                 -> f16
//           logits = hidden @ W2^T + b2                     -> f32 (d_out)
// ---------------------------------------------------------------------------
template <int ACT, typename OutT>
__global__ void __launch_bounds__(256)
gemm_wmma_kernel(const _Float16* __restrict__ A,  // [M,K]
                 const _Float16* __restrict__ W,  // [N,K]
                 const float* __restrict__ bias,  // [N]
                 OutT* __restrict__ out,          // [M,N]
                 int M, int N, int K) {
  const int lane = threadIdx.x & 31;
  const int wave = threadIdx.x >> 5;
  const int gw   = blockIdx.x * 8 + wave;
  const int npairs = N >> 5;
  const int m_base = (gw / npairs) << 4;
  const int n_base = (gw % npairs) << 5;
  if (m_base >= M) return;
  const int hi = lane >> 4;
  const int lm = lane & 15;

  const _Float16* __restrict__ a_row = A + (m_base + lm) * K;
  const _Float16* __restrict__ w0    = W + (n_base + lm) * K;
  const _Float16* __restrict__ w1    = W + (n_base + 16 + lm) * K;

  v8f c0 = {};
  v8f c1 = {};
#pragma unroll 8
  for (int kb = 0; kb < K; kb += 32) {
    v16h a  = load_a_h16(a_row, kb, hi);
    v16h b0 = load_b_h16(w0, kb, hi);
    v16h b1 = load_b_h16(w1, kb, hi);
    c0 = wmma_f16(a, b0, c0);
    c1 = wmma_f16(a, b1, c1);
  }

  const int n0 = n_base + lm;
  const int n1 = n0 + 16;
  const float bias0 = bias[n0];
  const float bias1 = bias[n1];
#pragma unroll
  for (int r = 0; r < 8; ++r) {
    const int m = m_base + r + 8 * hi;
    float v0 = c0[r] + bias0;
    float v1 = c1[r] + bias1;
    if (ACT == 1) {
      v0 = fmaxf(v0, 0.0f);
      v1 = fmaxf(v1, 0.0f);
    }
    out[m * N + n0] = (OutT)v0;
    out[m * N + n1] = (OutT)v1;
  }
}

// ---------------------------------------------------------------------------
// One RNN step on the sequential critical path:
//   h_cur = tanh(P[x_in[:,t]] + h_prev @ W_hh^T)
// The input projection + both biases are pre-folded into P [V,H] (f32), and
// gathered directly into the WMMA C accumulator (its element (r,lane) maps to
// row m_base + r + 8*hi, col n -- exactly the C/D layout).  Only the K=512
// recurrent GEMM remains in the serial chain.
// Rows with x_lens-1 == t also write into `last`.
// Grid: (B/16)*(H/32) = 1024 waves -> 128 blocks x 256 threads.
// ---------------------------------------------------------------------------
__global__ void __launch_bounds__(256)
rnn_step_kernel(const int* __restrict__ x_in,         // [B,S]
                const int* __restrict__ x_lens,       // [B]
                const float* __restrict__ P,          // [V,H] proj + biases
                const _Float16* __restrict__ whh16,   // [H,H]
                const _Float16* __restrict__ h_prev,  // [B,H]
                _Float16* __restrict__ h_cur,         // [B,H]
                _Float16* __restrict__ last,          // [B,H]
                int t) {
  const int lane = threadIdx.x & 31;
  const int wave = threadIdx.x >> 5;
  const int gw   = blockIdx.x * 8 + wave;
  const int m_base = (gw >> 4) << 4;  // 64 M-tiles
  const int n_base = (gw & 15) << 5;  // 16 N-pair-tiles (32 cols each)
  const int hi   = lane >> 4;
  const int lm   = lane & 15;
  const int n0   = n_base + lm;
  const int n1   = n0 + 16;

  // Initialize accumulators from the precomputed input-projection table.
  v8f c0, c1;
#pragma unroll
  for (int r = 0; r < 8; ++r) {
    const int m = m_base + r + 8 * hi;
    const int v = x_in[m * SS + t];
    c0[r] = P[v * HH + n0];
    c1[r] = P[v * HH + n1];
  }

  const _Float16* __restrict__ h_row = h_prev + (m_base + lm) * HH;
  const _Float16* __restrict__ wh0   = whh16 + n0 * HH;
  const _Float16* __restrict__ wh1   = whh16 + n1 * HH;

  // Recurrent projection: K over H from f16 h_prev.
#pragma unroll 8
  for (int kb = 0; kb < HH; kb += 32) {
    v16h a  = load_a_h16(h_row, kb, hi);
    v16h b0 = load_b_h16(wh0, kb, hi);
    v16h b1 = load_b_h16(wh1, kb, hi);
    c0 = wmma_f16(a, b0, c0);
    c1 = wmma_f16(a, b1, c1);
  }

  // Epilogue: biases already folded into P -> just tanh + store (f16).
#pragma unroll
  for (int r = 0; r < 8; ++r) {
    const int m = m_base + r + 8 * hi;
    const _Float16 v0 = (_Float16)tanhf(c0[r]);
    const _Float16 v1 = (_Float16)tanhf(c1[r]);
    h_cur[m * HH + n0] = v0;
    h_cur[m * HH + n1] = v1;
    if (x_lens[m] - 1 == t) {
      last[m * HH + n0] = v0;
      last[m * HH + n1] = v1;
    }
  }
}

// ---------------------------------------------------------------------------
// Host-side orchestration.
// Workspace layout:
//   f16: emb16 V*E | wih16 H*E | whh16 H*H | w1_16 H*H | w2_16 C*H
//        h16[2] 2*B*H | last16 B*H | hid16 B*H
//   f32: P V*H | bias2 H
// Total ~5.8 MB.  All regions naturally >=256B aligned.
// ---------------------------------------------------------------------------
extern "C" void kernel_launch(void* const* d_in, const int* in_sizes, int n_in,
                              void* d_out, int out_size, void* d_ws,
                              size_t ws_size, hipStream_t stream) {
  const int*   x_in   = (const int*)d_in[0];
  const int*   x_lens = (const int*)d_in[1];
  const float* emb    = (const float*)d_in[2];
  const float* W_ih   = (const float*)d_in[3];
  const float* b_ih   = (const float*)d_in[4];
  const float* W_hh   = (const float*)d_in[5];
  const float* b_hh   = (const float*)d_in[6];
  const float* W1     = (const float*)d_in[7];
  const float* b1     = (const float*)d_in[8];
  const float* W2     = (const float*)d_in[9];
  const float* b2     = (const float*)d_in[10];

  _Float16* p = (_Float16*)d_ws;
  _Float16* emb16 = p;  p += (size_t)VV * EE;
  _Float16* wih16 = p;  p += (size_t)HH * EE;
  _Float16* whh16 = p;  p += (size_t)HH * HH;
  _Float16* w1_16 = p;  p += (size_t)HH * HH;
  _Float16* w2_16 = p;  p += (size_t)CC * HH;
  _Float16* h16a  = p;  p += (size_t)BB * HH;
  _Float16* h16b  = p;  p += (size_t)BB * HH;
  _Float16* last  = p;  p += (size_t)BB * HH;
  _Float16* hid   = p;  p += (size_t)BB * HH;
  float* Pproj = (float*)p;
  float* bias2 = Pproj + (size_t)VV * HH;

  // --- One-time parallel prep (before the sequential chain) ---------------
  auto cvt = [&](const float* s, _Float16* d, int n) {
    cvt_f32_to_f16_kernel<<<(n / 4 + 255) / 256, 256, 0, stream>>>(s, d, n);
  };
  cvt(emb,  emb16, VV * EE);
  cvt(W_ih, wih16, HH * EE);
  cvt(W_hh, whh16, HH * HH);
  cvt(W1,   w1_16, HH * HH);
  cvt(W2,   w2_16, CC * HH);

  bias_fold_kernel<<<(HH + 255) / 256, 256, 0, stream>>>(b_ih, b_hh, bias2,
                                                         HH);

  // P = emb @ W_ih^T + (b_ih + b_hh): M=V=128, N=H=512, K=E=256 -> 128 waves.
  gemm_wmma_kernel<0, float><<<16, 256, 0, stream>>>(emb16, wih16, bias2,
                                                     Pproj, VV, HH, EE);

  // h0 = 0; last = 0 (f16 zero is all-zero bytes; memset is capturable).
  hipMemsetAsync(h16a, 0, (size_t)BB * HH * sizeof(_Float16), stream);
  hipMemsetAsync(last, 0, (size_t)BB * HH * sizeof(_Float16), stream);

  // --- 64 sequential RNN steps: 1024 waves = 128 blocks each --------------
  _Float16* hp[2] = {h16a, h16b};
  for (int t = 0; t < SS; ++t) {
    rnn_step_kernel<<<128, 256, 0, stream>>>(x_in, x_lens, Pproj, whh16,
                                             hp[t & 1], hp[(t + 1) & 1], last,
                                             t);
  }

  // --- MLP head ------------------------------------------------------------
  // hidden = relu(last @ W1^T + b1): M=1024, N=512, K=512 -> 1024 waves.
  gemm_wmma_kernel<1, _Float16><<<128, 256, 0, stream>>>(last, w1_16, b1, hid,
                                                         BB, HH, HH);
  // logits = hidden @ W2^T + b2: M=1024, N=32, K=512 -> 64 waves -> 8 blocks.
  gemm_wmma_kernel<0, float><<<8, 256, 0, stream>>>(hid, w2_16, b2,
                                                    (float*)d_out, BB, CC, HH);
}
